// FocusedDecoder_65103114272836
// MI455X (gfx1250) — matrix operand order repaired and checked by reading, not compiled, not run on hardware
//
#include <hip/hip_runtime.h>

typedef __attribute__((ext_vector_type(16))) _Float16 v16h;
typedef __attribute__((ext_vector_type(8)))  _Float16 v8h;
typedef __attribute__((ext_vector_type(8)))  float    v8f;

#define BS     8
#define DM     256
#define NH     8
#define DH     32
#define NL     6
#define DF     1024
#define SLEN   12800
#define NQ     60
#define NROW   (BS * NQ)        // 480
#define ROWCAP 512              // padded row capacity for q-tile padding to 64
#define SPLITS 5
#define SCHUNK (SLEN / SPLITS)  // 2560
#define TS     64
#define NT     (SCHUNK / TS)    // 40
#define QSCALE 0.1767766952966369f  // 1/sqrt(32)

// ---------------------------------------------------------------- WMMA helpers

__device__ __forceinline__ v8f wmma32(v16h a, v16h b, v8f c) {
  return __builtin_amdgcn_wmma_f32_16x16x32_f16(false, a, false, b, (short)0, c,
                                                false, false);
}

// A fragment 16x32 (MxK) from row-major f16, p -> element [m0][k0], stride lda.
// lanes 0-15: row=lane, K 0..7 & 16..23 ; lanes 16-31: row=lane-16, K 8..15 & 24..31
__device__ __forceinline__ v16h ldA(const _Float16* p, int lda, int lane) {
  int r = lane & 15, hf = lane >> 4;
  const _Float16* q = p + r * lda + hf * 8;
  v8h lo = *(const v8h*)(q);
  v8h hi = *(const v8h*)(q + 16);
  v16h out;
#pragma unroll
  for (int i = 0; i < 8; ++i) { out[i] = lo[i]; out[i + 8] = hi[i]; }
  return out;
}

// B fragment 32x16 (KxN) where source is stored row-major [n][k] (e.g. weight
// [N,K] so B[k][n]=W[n][k]); p -> element [n0][k0], stride ldb. Same K order as ldA.
__device__ __forceinline__ v16h ldB(const _Float16* p, int ldb, int lane) {
  const _Float16* q = p + (lane & 15) * ldb + (lane >> 4) * 8;
  v8h lo = *(const v8h*)(q);
  v8h hi = *(const v8h*)(q + 16);
  v16h out;
#pragma unroll
  for (int i = 0; i < 8; ++i) { out[i] = lo[i]; out[i + 8] = hi[i]; }
  return out;
}

// ------------------------------------------------------------ generic WMMA GEMM
// C[M,N] = alpha * (A[M,K] @ W[N,K]^T + bias), optional relu, f32 and/or f16 out.
// grid = (N/128, M/16), block = 256 (8 waves, one 16x16 tile per wave).
__global__ __launch_bounds__(256) void gemm16(
    const _Float16* __restrict__ A, const _Float16* __restrict__ W,
    const float* __restrict__ bias, float* __restrict__ Cf,
    _Float16* __restrict__ Ch, int N, int K, float alpha, int relu) {
  int lane = threadIdx.x & 31, wave = threadIdx.x >> 5;
  int n0 = blockIdx.x * 128 + wave * 16;
  int m0 = blockIdx.y * 16;
  v8f acc = {};
  for (int k0 = 0; k0 < K; k0 += 32) {
    v16h a = ldA(A + (size_t)m0 * K + k0, K, lane);
    v16h b = ldB(W + (size_t)n0 * K + k0, K, lane);
    acc = wmma32(a, b, acc);
  }
  int n = n0 + (lane & 15);
  int hf = lane >> 4;
  float bb = bias ? bias[n] : 0.0f;
#pragma unroll
  for (int i = 0; i < 8; ++i) {
    int m = m0 + i + 8 * hf;
    float v = alpha * (acc[i] + bb);
    if (relu) v = fmaxf(v, 0.0f);
    if (Cf) Cf[(size_t)m * N + n] = v;
    if (Ch) Ch[(size_t)m * N + n] = (_Float16)v;
  }
}

// ------------------------------------------------------- self-attention (B=8,H=8)
// grid = 64 (b,h), block = 64 (2 waves). Q/K/V precomputed f16 [ROWCAP,256].
__global__ __launch_bounds__(64) void sa_attn(
    const _Float16* __restrict__ qh, const _Float16* __restrict__ kh,
    const _Float16* __restrict__ vh, _Float16* __restrict__ outh) {
  int b = blockIdx.x >> 3, h = blockIdx.x & 7;
  int tid = threadIdx.x, lane = tid & 31, wave = tid >> 5;
  __shared__ float    sc[64 * 64] __attribute__((aligned(32)));
  __shared__ _Float16 P[64 * 64] __attribute__((aligned(32)));
  __shared__ _Float16 vt[DH * 64] __attribute__((aligned(32)));  // [d][s]
  __shared__ float    rowl[64];

  {  // stage V transposed in LDS
    int s = tid;
    const _Float16* vp = vh + (size_t)(b * NQ + s) * DM + h * DH;
#pragma unroll
    for (int d = 0; d < DH; ++d) vt[d * 64 + s] = vp[d];
  }
  __syncthreads();
  // scores (q pre-scaled by 1/sqrt(32)); 16 tiles, 8 per wave
  for (int j = 0; j < 8; ++j) {
    int id = wave * 8 + j;
    int qt = id >> 2, st = id & 3;
    v16h a = ldA(qh + (size_t)(b * NQ + qt * 16) * DM + h * DH, DM, lane);
    v16h bf = ldB(kh + (size_t)(b * NQ + st * 16) * DM + h * DH, DM, lane);
    v8f acc = {};
    acc = wmma32(a, bf, acc);
    int hf = lane >> 4, ln = lane & 15;
#pragma unroll
    for (int i = 0; i < 8; ++i)
      sc[(qt * 16 + i + 8 * hf) * 64 + st * 16 + ln] = acc[i];
  }
  __syncthreads();
  {  // full softmax per row (keep P unnormalized; divide in PV epilogue)
    int q = tid;
    float m = -1e30f;
    for (int c = 0; c < 64; ++c) {
      float v = (c < NQ) ? sc[q * 64 + c] : -1e9f;
      m = fmaxf(m, v);
    }
    float l = 0.0f;
    for (int c = 0; c < 64; ++c) {
      float v = (c < NQ) ? sc[q * 64 + c] : -1e9f;
      float p = __expf(v - m);
      P[q * 64 + c] = (_Float16)p;
      l += p;
    }
    rowl[q] = l;
  }
  __syncthreads();
  // out = P @ V ; 8 tiles, 4 per wave
  for (int j = 0; j < 4; ++j) {
    int id = wave * 4 + j;
    int qt = id >> 1, dt = id & 1;
    v8f acc = {};
    for (int k = 0; k < 2; ++k) {
      v16h a = ldA(P + qt * 16 * 64 + k * 32, 64, lane);
      v16h bf = ldB(vt + dt * 16 * 64 + k * 32, 64, lane);
      acc = wmma32(a, bf, acc);
    }
    int hf = lane >> 4, ln = lane & 15;
#pragma unroll
    for (int i = 0; i < 8; ++i) {
      int q = qt * 16 + i + 8 * hf;
      if (q < NQ)
        outh[(size_t)(b * NQ + q) * DM + h * DH + dt * 16 + ln] =
            (_Float16)(acc[i] / rowl[q]);
    }
  }
}

// ---------------------------------------------- fused cross-attention (split-S)
// grid = B*H*SPLITS = 320, block = 256 (8 waves). Projects K/V tiles on the fly
// from L2-resident kv/src f16, online masked softmax, PV; emits flash partials.
__global__ __launch_bounds__(256) void ca_fused(
    const _Float16* __restrict__ kv16, const _Float16* __restrict__ sx16,
    const _Float16* __restrict__ qh, const _Float16* __restrict__ wkv,
    const unsigned char* __restrict__ mask, float* __restrict__ pout,
    float* __restrict__ pm, float* __restrict__ pl) {
  int bid = blockIdx.x;
  int sp = bid % SPLITS;
  int h = (bid / SPLITS) & 7;
  int b = bid / (SPLITS * NH);
  int tid = threadIdx.x, lane = tid & 31, wave = tid >> 5;

  __shared__ _Float16 kh[TS * DH] __attribute__((aligned(32)));  // [s][d]
  __shared__ _Float16 vt[DH * TS] __attribute__((aligned(32)));  // [d][s]
  __shared__ float    sc[64 * TS] __attribute__((aligned(32)));  // [q][s]
  __shared__ _Float16 P[64 * TS] __attribute__((aligned(32)));
  __shared__ float    rowm[64], rowl[64], rowa[64];

  if (tid < 64) { rowm[tid] = -1e30f; rowl[tid] = 0.0f; }

  const _Float16* wk  = wkv + (size_t)(DM + h * DH) * DM;      // K weight rows
  const _Float16* wv  = wkv + (size_t)(2 * DM + h * DH) * DM;  // V weight rows
  const _Float16* Akv = kv16 + (size_t)b * SLEN * DM;
  const _Float16* Asx = sx16 + (size_t)b * SLEN * DM;

  int qt_o = wave >> 1, dt_o = wave & 1;  // this wave's output tile
  v16h qfrag = ldA(qh + ((size_t)(b * NQ) + qt_o * 16) * DM + h * DH, DM, lane);
  v8f oacc = {};
  __syncthreads();

  int s_base = sp * SCHUNK;
  for (int t = 0; t < NT; ++t) {
    int s0 = s_base + t * TS;
    // ---- phase 1: project K and V 64x32 tiles into LDS (16 wmma-tiles)
    for (int j = 0; j < 2; ++j) {
      int id = wave * 2 + j;
      int isv = id >> 3;
      int rt = (id >> 1) & 3, ct = id & 1;
      const _Float16* A = isv ? Asx : Akv;
      const _Float16* W = isv ? wv : wk;
      v8f acc = {};
      for (int k0 = 0; k0 < DM; k0 += 32) {
        v16h a = ldA(A + (size_t)(s0 + rt * 16) * DM + k0, DM, lane);
        v16h bf = ldB(W + ct * 16 * DM + k0, DM, lane);
        acc = wmma32(a, bf, acc);
      }
      int hf = lane >> 4, ln = lane & 15;
      if (!isv) {
#pragma unroll
        for (int i = 0; i < 8; ++i)
          kh[(rt * 16 + i + 8 * hf) * DH + ct * 16 + ln] = (_Float16)acc[i];
      } else {
#pragma unroll
        for (int i = 0; i < 8; ++i)
          vt[(ct * 16 + ln) * TS + rt * 16 + 8 * hf + i] = (_Float16)acc[i];
      }
    }
    __syncthreads();
    // ---- phase 2: scores = (q/sqrt(d)) @ K^T  (16 tiles, qt matches qt_o)
    for (int j = 0; j < 2; ++j) {
      int id = wave * 2 + j;
      int qt = id >> 2, st = id & 3;
      v16h bf = ldB(kh + st * 16 * DH, DH, lane);
      v8f acc = {};
      acc = wmma32(qfrag, bf, acc);
      int hf = lane >> 4, ln = lane & 15;
#pragma unroll
      for (int i = 0; i < 8; ++i)
        sc[(qt * 16 + i + 8 * hf) * TS + st * 16 + ln] = acc[i];
      (void)qt;
    }
    __syncthreads();
    // ---- phase 3: masked online softmax (one thread per query row)
    if (tid < 64) {
      int q = tid;
      float m_old = rowm[q];
      float mx = m_old;
      if (q < NQ) {
        const unsigned char* mrow = mask + (size_t)q * SLEN + s0;
        for (int c = 0; c < TS; ++c) {
          float v = sc[q * TS + c];
          if (mrow[c]) v = -1e9f;
          sc[q * TS + c] = v;
          mx = fmaxf(mx, v);
        }
      } else {
        for (int c = 0; c < TS; ++c) mx = fmaxf(mx, sc[q * TS + c]);
      }
      float a = __expf(m_old - mx);
      float l = rowl[q] * a;
      for (int c = 0; c < TS; ++c) {
        float p = __expf(sc[q * TS + c] - mx);
        P[q * TS + c] = (_Float16)p;
        l += p;
      }
      rowm[q] = mx; rowl[q] = l; rowa[q] = a;
    }
    __syncthreads();
    // ---- phase 4: rescale running out and accumulate P @ V
    {
      int hf = lane >> 4;
#pragma unroll
      for (int i = 0; i < 8; ++i) oacc[i] *= rowa[qt_o * 16 + i + 8 * hf];
      for (int k = 0; k < 2; ++k) {
        v16h a = ldA(P + qt_o * 16 * TS + k * 32, TS, lane);
        v16h bf = ldB(vt + dt_o * 16 * TS + k * 32, TS, lane);
        oacc = wmma32(a, bf, oacc);
      }
    }
    __syncthreads();
  }
  // ---- write flash partials (unnormalized O, running m, running l)
  {
    int hf = lane >> 4, ln = lane & 15;
    float* po = pout + (size_t)bid * (NQ * DH);
#pragma unroll
    for (int i = 0; i < 8; ++i) {
      int q = qt_o * 16 + i + 8 * hf;
      if (q < NQ) po[q * DH + dt_o * 16 + ln] = oacc[i];
    }
  }
  if (tid < NQ) {
    pm[(size_t)bid * NQ + tid] = rowm[tid];
    pl[(size_t)bid * NQ + tid] = rowl[tid];
  }
}

// ------------------------------------------------ combine split partials (+bv)
__global__ __launch_bounds__(256) void ca_combine(
    const float* __restrict__ pout, const float* __restrict__ pm,
    const float* __restrict__ pl, const float* __restrict__ bv,
    _Float16* __restrict__ outh) {
  int bh = blockIdx.x;  // b*8 + h
  int b = bh >> 3, h = bh & 7;
  for (int idx = threadIdx.x; idx < NQ * DH; idx += blockDim.x) {
    int q = idx / DH, d = idx - q * DH;
    float M = -1e30f;
    for (int j = 0; j < SPLITS; ++j)
      M = fmaxf(M, pm[((size_t)bh * SPLITS + j) * NQ + q]);
    float L = 0.0f, O = 0.0f;
    for (int j = 0; j < SPLITS; ++j) {
      size_t pidx = (size_t)bh * SPLITS + j;
      float w = __expf(pm[pidx * NQ + q] - M);
      L += pl[pidx * NQ + q] * w;
      O += pout[pidx * (NQ * DH) + idx] * w;
    }
    float val = O / L + bv[h * DH + d];  // V-bias exactly (softmax sums to 1)
    outh[(size_t)(b * NQ + q) * DM + h * DH + d] = (_Float16)val;
  }
}

// ------------------------------------- [B,256,S] -> [B,S,256] f16 (kv and src)
__global__ __launch_bounds__(256) void build_kv(
    const float* __restrict__ srcs, const float* __restrict__ pos,
    _Float16* __restrict__ kv16, _Float16* __restrict__ sx16) {
  __shared__ float t1[32][33];
  __shared__ float t2[32][33];
  int b = blockIdx.z, d0 = blockIdx.y * 32, s0 = blockIdx.x * 32;
  int tx = threadIdx.x, ty = threadIdx.y;  // 32 x 8
  for (int j = 0; j < 4; ++j) {
    int dd = ty + j * 8;
    size_t gi = ((size_t)b * DM + d0 + dd) * SLEN + s0 + tx;
    float sv = srcs[gi], pv = pos[gi];
    t2[dd][tx] = sv;
    t1[dd][tx] = sv + pv;
  }
  __syncthreads();
  for (int j = 0; j < 4; ++j) {
    int ss = ty + j * 8;
    size_t go = ((size_t)b * SLEN + s0 + ss) * DM + d0 + tx;
    kv16[go] = (_Float16)t1[tx][ss];
    sx16[go] = (_Float16)t2[tx][ss];
  }
}

// -------------------------------------------------------------- small helpers
__global__ void cvt_f16(const float* __restrict__ x, _Float16* __restrict__ y,
                        int n) {
  int i = blockIdx.x * blockDim.x + threadIdx.x;
  if (i < n) y[i] = (_Float16)x[i];
}

__global__ void tgt_init(const float* __restrict__ qe, float* __restrict__ tgt) {
  int i = blockIdx.x * 256 + threadIdx.x;  // over NROW*DM
  int r = i >> 8, d = i & 255;
  tgt[i] = qe[(r % NQ) * 2 * DM + DM + d];
}

__global__ void row_add_cvt(const float* __restrict__ x,
                            const float* __restrict__ qe,
                            _Float16* __restrict__ y) {
  int i = blockIdx.x * 256 + threadIdx.x;
  int r = i >> 8, d = i & 255;
  float v = x[i];
  if (qe) v += qe[(r % NQ) * 2 * DM + d];
  y[i] = (_Float16)v;
}

__global__ __launch_bounds__(256) void ln_residual(
    float* __restrict__ tgt, const float* __restrict__ delta,
    const float* __restrict__ g, const float* __restrict__ be) {
  int r = blockIdx.x, d = threadIdx.x;
  __shared__ float red[256];
  float x = tgt[(size_t)r * DM + d] + delta[(size_t)r * DM + d];
  red[d] = x;
  __syncthreads();
  for (int s = 128; s > 0; s >>= 1) {
    if (d < s) red[d] += red[d + s];
    __syncthreads();
  }
  float mu = red[0] * (1.0f / DM);
  __syncthreads();
  float diff = x - mu;
  red[d] = diff * diff;
  __syncthreads();
  for (int s = 128; s > 0; s >>= 1) {
    if (d < s) red[d] += red[d + s];
    __syncthreads();
  }
  float var = red[0] * (1.0f / DM);
  tgt[(size_t)r * DM + d] = diff * rsqrtf(var + 1e-5f) * g[d] + be[d];
}

// ------------------------------------------------------------------- launcher
extern "C" void kernel_launch(void* const* d_in, const int* in_sizes, int n_in,
                              void* d_out, int out_size, void* d_ws,
                              size_t ws_size, hipStream_t stream) {
  (void)in_sizes; (void)n_in; (void)out_size; (void)ws_size;
  const float* srcs = (const float*)d_in[0];
  const float* pos = (const float*)d_in[1];
  const float* qe = (const float*)d_in[2];
  const unsigned char* mask = (const unsigned char*)d_in[3];
  const float* sa_w_in = (const float*)d_in[4];
  const float* sa_b_in = (const float*)d_in[5];
  const float* sa_w_out = (const float*)d_in[6];
  const float* sa_b_out = (const float*)d_in[7];
  const float* ca_w_in = (const float*)d_in[8];
  const float* ca_b_in = (const float*)d_in[9];
  const float* ca_w_out = (const float*)d_in[10];
  const float* ca_b_out = (const float*)d_in[11];
  const float* w1 = (const float*)d_in[12];
  const float* b1 = (const float*)d_in[13];
  const float* w2 = (const float*)d_in[14];
  const float* b2 = (const float*)d_in[15];
  const float* ln1_g = (const float*)d_in[16];
  const float* ln1_b = (const float*)d_in[17];
  const float* ln2_g = (const float*)d_in[18];
  const float* ln2_b = (const float*)d_in[19];
  const float* ln3_g = (const float*)d_in[20];
  const float* ln3_b = (const float*)d_in[21];

  char* base = (char*)d_ws;
  size_t off = 0;
  auto alloc = [&](size_t bytes) {
    char* p = base + off;
    off += (bytes + 255) & ~(size_t)255;
    return p;
  };
  _Float16* kv16 = (_Float16*)alloc((size_t)BS * SLEN * DM * 2);
  _Float16* sx16 = (_Float16*)alloc((size_t)BS * SLEN * DM * 2);
  _Float16* sa_w_in16 = (_Float16*)alloc((size_t)NL * 3 * DM * DM * 2);
  _Float16* sa_w_out16 = (_Float16*)alloc((size_t)NL * DM * DM * 2);
  _Float16* ca_w_in16 = (_Float16*)alloc((size_t)NL * 3 * DM * DM * 2);
  _Float16* ca_w_out16 = (_Float16*)alloc((size_t)NL * DM * DM * 2);
  _Float16* w1_16 = (_Float16*)alloc((size_t)NL * DF * DM * 2);
  _Float16* w2_16 = (_Float16*)alloc((size_t)NL * DM * DF * 2);
  float* tgt = (float*)alloc((size_t)ROWCAP * DM * 4);
  float* resid = (float*)alloc((size_t)ROWCAP * DM * 4);
  _Float16* x16 = (_Float16*)alloc((size_t)ROWCAP * DM * 2);
  _Float16* t16 = (_Float16*)alloc((size_t)ROWCAP * DM * 2);
  _Float16* qh16 = (_Float16*)alloc((size_t)ROWCAP * DM * 2);
  _Float16* kh16 = (_Float16*)alloc((size_t)ROWCAP * DM * 2);
  _Float16* vh16 = (_Float16*)alloc((size_t)ROWCAP * DM * 2);
  _Float16* head16 = (_Float16*)alloc((size_t)ROWCAP * DM * 2);
  _Float16* qca16 = (_Float16*)alloc((size_t)ROWCAP * DM * 2);
  _Float16* ffh16 = (_Float16*)alloc((size_t)ROWCAP * DF * 2);
  float* pout = (float*)alloc((size_t)BS * NH * SPLITS * NQ * DH * 4);
  float* pm = (float*)alloc((size_t)BS * NH * SPLITS * NQ * 4);
  float* pl = (float*)alloc((size_t)BS * NH * SPLITS * NQ * 4);

  auto CVT = [&](const float* s, _Float16* d, size_t n) {
    cvt_f16<<<dim3((unsigned)((n + 255) / 256)), 256, 0, stream>>>(s, d, (int)n);
  };
  CVT(sa_w_in, sa_w_in16, (size_t)NL * 3 * DM * DM);
  CVT(sa_w_out, sa_w_out16, (size_t)NL * DM * DM);
  CVT(ca_w_in, ca_w_in16, (size_t)NL * 3 * DM * DM);
  CVT(ca_w_out, ca_w_out16, (size_t)NL * DM * DM);
  CVT(w1, w1_16, (size_t)NL * DF * DM);
  CVT(w2, w2_16, (size_t)NL * DM * DF);
  build_kv<<<dim3(SLEN / 32, DM / 32, BS), dim3(32, 8), 0, stream>>>(srcs, pos,
                                                                     kv16, sx16);
  tgt_init<<<NROW, 256, 0, stream>>>(qe, tgt);

  for (int i = 0; i < NL; ++i) {
    const _Float16* swi = sa_w_in16 + (size_t)i * 3 * DM * DM;
    const _Float16* cwi = ca_w_in16 + (size_t)i * 3 * DM * DM;
    dim3 g1(DM / 128, NROW / 16);
    // ---- self attention
    row_add_cvt<<<NROW, 256, 0, stream>>>(tgt, qe, x16);
    row_add_cvt<<<NROW, 256, 0, stream>>>(tgt, nullptr, t16);
    gemm16<<<g1, 256, 0, stream>>>(x16, swi, sa_b_in + i * 3 * DM, nullptr,
                                   qh16, DM, DM, QSCALE, 0);
    gemm16<<<g1, 256, 0, stream>>>(x16, swi + DM * DM,
                                   sa_b_in + i * 3 * DM + DM, nullptr, kh16, DM,
                                   DM, 1.0f, 0);
    gemm16<<<g1, 256, 0, stream>>>(t16, swi + 2 * DM * DM,
                                   sa_b_in + i * 3 * DM + 2 * DM, nullptr, vh16,
                                   DM, DM, 1.0f, 0);
    sa_attn<<<BS * NH, 64, 0, stream>>>(qh16, kh16, vh16, head16);
    gemm16<<<g1, 256, 0, stream>>>(head16, sa_w_out16 + (size_t)i * DM * DM,
                                   sa_b_out + i * DM, resid, nullptr, DM, DM,
                                   1.0f, 0);
    ln_residual<<<NROW, 256, 0, stream>>>(tgt, resid, ln2_g + i * DM,
                                          ln2_b + i * DM);
    // ---- cross attention (fused K/V projection + flash softmax)
    row_add_cvt<<<NROW, 256, 0, stream>>>(tgt, qe, x16);
    gemm16<<<g1, 256, 0, stream>>>(x16, cwi, ca_b_in + i * 3 * DM, nullptr,
                                   qca16, DM, DM, QSCALE, 0);
    ca_fused<<<BS * NH * SPLITS, 256, 0, stream>>>(kv16, sx16, qca16, cwi, mask,
                                                   pout, pm, pl);
    ca_combine<<<BS * NH, 256, 0, stream>>>(pout, pm, pl,
                                            ca_b_in + i * 3 * DM + 2 * DM,
                                            head16);
    gemm16<<<g1, 256, 0, stream>>>(head16, ca_w_out16 + (size_t)i * DM * DM,
                                   ca_b_out + i * DM, resid, nullptr, DM, DM,
                                   1.0f, 0);
    ln_residual<<<NROW, 256, 0, stream>>>(tgt, resid, ln1_g + i * DM,
                                          ln1_b + i * DM);
    // ---- FFN
    row_add_cvt<<<NROW, 256, 0, stream>>>(tgt, nullptr, t16);
    gemm16<<<dim3(DF / 128, NROW / 16), 256, 0, stream>>>(
        t16, w1_16 + (size_t)i * DF * DM, b1 + i * DF, nullptr, ffh16, DF, DM,
        1.0f, 1);
    gemm16<<<g1, 256, 0, stream>>>(ffh16, w2_16 + (size_t)i * DM * DF,
                                   b2 + i * DM, resid, nullptr, DM, DF, 1.0f,
                                   0);
    ln_residual<<<NROW, 256, 0, stream>>>(tgt, resid, ln3_g + i * DM,
                                          ln3_b + i * DM);
  }
  hipMemcpyAsync(d_out, tgt, (size_t)NROW * DM * sizeof(float),
                 hipMemcpyDeviceToDevice, stream);
}
